// Fusion_5617817223437
// MI455X (gfx1250) — matrix operand order repaired and checked by reading, not compiled, not run on hardware
//
#include <hip/hip_runtime.h>

// Problem constants (match reference)
#define KP     800000
#define N3D_   100000
#define NTILE  (KP / 16)      // 50000 position tiles of 16
#define FILLV  -9999.0f

typedef __attribute__((ext_vector_type(16))) _Float16     v16h;
typedef __attribute__((ext_vector_type(8)))  float        v8f;
typedef __attribute__((ext_vector_type(2)))  __fp16       v2fp;  // cvt_pkrtz return type
typedef __attribute__((ext_vector_type(4)))  unsigned int v4u;

union Frag { v16h h; unsigned int u[8]; v4u q[2]; };

__device__ __forceinline__ unsigned int pk2(float a, float b) {
  union { v2fp h; unsigned int u; } cv;
  cv.h = __builtin_amdgcn_cvt_pkrtz(a, b);   // low = a, high = b
  return cv.u;
}

// Pack pair to f16 with fused ReLU applied on the f16 side:
// cvt_pkrtz output is known-canonical, so max lowers to a single
// v_pk_max_num_f16 (no canonicalize) -> 2 VALU ops per pair total.
__device__ __forceinline__ unsigned int pk2r(float a, float b) {
  v2fp pk = __builtin_amdgcn_cvt_pkrtz(a, b);
  v2fp z  = {(__fp16)0.0f, (__fp16)0.0f};
  v2fp r  = __builtin_elementwise_max(pk, z);
  union { v2fp h; unsigned int u; } cv;
  cv.h = r;
  return cv.u;
}

__device__ __forceinline__ v8f wmma_f16(const Frag& a, const Frag& b, v8f c) {
  // D = A(16x32 f16) * B(32x16 f16) + C(16x16 f32)
  return __builtin_amdgcn_wmma_f32_16x16x32_f16(
      false, a.h, false, b.h, (short)0, c, false, false);
}

// Native CDNA5 float atomic max (device scope), no-return form -> STOREcnt.
__device__ __forceinline__ void atomic_max_f32(float* addr, float v) {
  asm volatile("global_atomic_max_num_f32 %0, %1, off scope:SCOPE_DEV"
               :: "v"((unsigned long long)addr), "v"(v)
               : "memory");
}

// ---------------- weight prep ------------------------------------------------
// Channel-permutation trick: K-slot k (physical B layout position) sources
// original channel 16*((k>>3)&1) + 8*(k>>4) + (k&7), so every D(f32)->B(f16)
// transition is a pure per-lane pack (no cross-lane traffic).
// mode 0: identity (layer-1 input); mode 1: permuted 32-wide chunk at col 0;
// mode 2: second K-chunk (cols 32..39 on its lo-entry slots, rest don't-care).
__device__ __forceinline__ int perm_col(int k, int mode) {
  if (mode == 0) return k;
  if (mode == 1) return 16 * ((k >> 3) & 1) + 8 * (k >> 4) + (k & 7);
  return ((k >> 3) & 1) ? 63 : (32 + 8 * (k >> 4) + (k & 7));
}

__device__ __forceinline__ float wval(const float* W, const float* b,
                                      int cin, int cout, int ch, int k) {
  if (ch >= cout) return 0.0f;
  if (k < cin)    return W[ch * cin + k];
  if (k == cin)   return b[ch];        // bias folded as extra input column
  return 0.0f;
}

// 13 A-tiles: [0,1]=L1(3+b->18), [2..4]=L2(18+b->36), [5..10]=L3(36+b->36,
// 3 out-tiles x 2 K-chunks), [11,12]=L4(36+b->1, 2 K-chunks).
// A-fragment layout (16-bit A 16x32): lane l -> row m=l&15, half h=l>>4;
// VGPR v holds K pair starting at k0 = 16*h + 2*v.
// Stored as frags[tile*256 + lane*8 + v] so each lane loads 2x b128.
__global__ void prep_kernel(const float* __restrict__ W1, const float* __restrict__ b1,
                            const float* __restrict__ W2, const float* __restrict__ b2,
                            const float* __restrict__ W3, const float* __restrict__ b3,
                            const float* __restrict__ W4, const float* __restrict__ b4,
                            unsigned int* __restrict__ frags) {
  int tile = blockIdx.x;                 // 0..12
  int tid  = threadIdx.x;                // 0..255
  int v = tid >> 5, lane = tid & 31;
  int m = lane & 15, h = lane >> 4;

  int ob, mode;
  const float *W, *b; int cin, cout;
  if (tile < 2)       { W = W1; b = b1; cin = 3;  cout = 18; ob = tile * 16;       mode = 0; }
  else if (tile < 5)  { W = W2; b = b2; cin = 18; cout = 36; ob = (tile - 2) * 16; mode = 1; }
  else if (tile < 11) { int t = tile - 5;
                        W = W3; b = b3; cin = 36; cout = 36; ob = (t >> 1) * 16;
                        mode = (t & 1) ? 2 : 1; }
  else                { W = W4; b = b4; cin = 36; cout = 1;  ob = 0;
                        mode = (tile == 11) ? 1 : 2; }

  int ch = ob + m;
  int k0 = 16 * h + 2 * v;                          // physical K slot in chunk
  frags[tile * 256 + lane * 8 + v] =
      pk2(wval(W, b, cin, cout, ch, perm_col(k0,     mode)),
          wval(W, b, cin, cout, ch, perm_col(k0 + 1, mode)));
}

__global__ void init_kernel(float* __restrict__ out, int n) {
  int i = blockIdx.x * blockDim.x + threadIdx.x;
  if (i < n) out[i] = FILLV;
}

// ---------------- main fused MLP + atomic-max scatter -----------------------
__global__ void __launch_bounds__(256)
mlp_scatter_max_kernel(const float* __restrict__ x, const int* __restrict__ idx0,
                       const unsigned int* __restrict__ frags,
                       float* __restrict__ out) {
  const int  lane = threadIdx.x & 31;
  const int  wave = (blockIdx.x * blockDim.x + threadIdx.x) >> 5;
  const int  nwav = (gridDim.x * blockDim.x) >> 5;
  const int  m    = lane & 15;
  const bool hi   = lane >= 16;

  // Preload all 13 A fragments (weights, f16, pre-swizzled): 26x b128 loads.
  Frag A[13];
#pragma unroll
  for (int t = 0; t < 13; ++t) {
    const v4u* fp = (const v4u*)(frags + t * 256 + lane * 8);
    A[t].q[0] = fp[0];
    A[t].q[1] = fp[1];
  }

  const v8f zf = {0.f, 0.f, 0.f, 0.f, 0.f, 0.f, 0.f, 0.f};
  const unsigned int one_sel = hi ? 0u : 0x3C00u;   // f16 1.0 in low half, lo lanes

  // Software-pipelined input loads (branch-free: last-iteration lookahead is
  // clamped to a valid tile; the values are dead once the loop exits).
  int t = wave;
  if (t >= NTILE) return;
  int p = t * 16 + m;
  float x0 = x[p], x1 = x[KP + p], x2 = x[2 * KP + p];
  int   i3 = idx0[p];

  for (; t < NTILE; ) {
    const int tn  = t + nwav;
    const int tc  = (tn < NTILE) ? tn : t;          // uniform clamp, no EXEC ops
    const int pn  = tc * 16 + m;
    const float n0 = x[pn], n1 = x[KP + pn], n2 = x[2 * KP + pn];
    const int   ni = idx0[pn];
    __builtin_prefetch(x + pn + nwav * 16, 0, 0);   // iter+2 hint (speculative)

    // Layer-1 input fragment: lo lanes K0..2 = x, K3 = 1.0 (bias); rest 0.
    Frag B;
    B.u[0] = hi ? 0u : pk2(x0, x1);
    B.u[1] = hi ? 0u : pk2(x2, 1.0f);
#pragma unroll
    for (int v = 2; v < 8; ++v) B.u[v] = 0u;

    // L1: 3(+bias)->18 (padded to 32 out channels, 2 tiles)
    v8f d0 = wmma_f16(A[0], B, zf);
    v8f d1 = wmma_f16(A[1], B, zf);

    // L1->L2: pure local pack; pad channels are constants, bias slot -> 1.0.
    Frag B2;
    B2.u[0] = pk2r(d0[0], d0[1]); B2.u[1] = pk2r(d0[2], d0[3]);
    B2.u[2] = pk2r(d0[4], d0[5]); B2.u[3] = pk2r(d0[6], d0[7]);
    B2.u[4] = pk2r(d1[0], d1[1]);
    B2.u[5] = one_sel; B2.u[6] = 0u; B2.u[7] = 0u;

    // L2: 18(+bias)->36 (3 out tiles)
    v8f e0 = wmma_f16(A[2], B2, zf);
    v8f e1 = wmma_f16(A[3], B2, zf);
    v8f e2 = wmma_f16(A[4], B2, zf);

    // L2->L3: chunk0 from e0/e1 (all real), chunk1 from e2.
    Frag Bc0, Bc1;
    Bc0.u[0] = pk2r(e0[0], e0[1]); Bc0.u[1] = pk2r(e0[2], e0[3]);
    Bc0.u[2] = pk2r(e0[4], e0[5]); Bc0.u[3] = pk2r(e0[6], e0[7]);
    Bc0.u[4] = pk2r(e1[0], e1[1]); Bc0.u[5] = pk2r(e1[2], e1[3]);
    Bc0.u[6] = pk2r(e1[4], e1[5]); Bc0.u[7] = pk2r(e1[6], e1[7]);
    Bc1.u[0] = pk2r(e2[0], e2[1]); Bc1.u[1] = pk2r(e2[2], e2[3]);
    Bc1.u[2] = one_sel;            // ch36 bias slot; ch37 = 0
    Bc1.u[3] = 0u;
#pragma unroll
    for (int v = 4; v < 8; ++v) Bc1.u[v] = 0u;

    // L3: 36(+bias)->36; two chained WMMAs per out tile
    v8f f0 = wmma_f16(A[5], Bc0, zf);  f0 = wmma_f16(A[6],  Bc1, f0);
    v8f f1 = wmma_f16(A[7], Bc0, zf);  f1 = wmma_f16(A[8],  Bc1, f1);
    v8f f2 = wmma_f16(A[9], Bc0, zf);  f2 = wmma_f16(A[10], Bc1, f2);

    // L3->L4: same pack pattern (f2 rows 4+ are pad-zero channels).
    Frag Bf0, Bf1;
    Bf0.u[0] = pk2r(f0[0], f0[1]); Bf0.u[1] = pk2r(f0[2], f0[3]);
    Bf0.u[2] = pk2r(f0[4], f0[5]); Bf0.u[3] = pk2r(f0[6], f0[7]);
    Bf0.u[4] = pk2r(f1[0], f1[1]); Bf0.u[5] = pk2r(f1[2], f1[3]);
    Bf0.u[6] = pk2r(f1[4], f1[5]); Bf0.u[7] = pk2r(f1[6], f1[7]);
    Bf1.u[0] = pk2r(f2[0], f2[1]); Bf1.u[1] = pk2r(f2[2], f2[3]);
    Bf1.u[2] = one_sel;            // ch36 bias slot (b4); ch37 = 0
    Bf1.u[3] = 0u;
#pragma unroll
    for (int v = 4; v < 8; ++v) Bf1.u[v] = 0u;

    // L4: 36(+bias)->1 as WMMA; y for this lane's position is D row 0,
    // i.e. g[0] on lo lanes (bias b4 already folded in).
    v8f g = wmma_f16(A[11], Bf0, zf);
    g = wmma_f16(A[12], Bf1, g);

    if (!hi) {
      atomic_max_f32(out + i3, g[0]);
    }

    x0 = n0; x1 = n1; x2 = n2; i3 = ni;
    t = tn;
  }
}

extern "C" void kernel_launch(void* const* d_in, const int* in_sizes, int n_in,
                              void* d_out, int out_size, void* d_ws, size_t ws_size,
                              hipStream_t stream) {
  (void)in_sizes; (void)n_in; (void)out_size; (void)ws_size;
  const float* x    = (const float*)d_in[0];
  // d_in[1] (T_out) unused: scatter+maxpool collapses to per-row max.
  const int*   tidx = (const int*)d_in[2];        // first K entries = idx0
  const float* W1 = (const float*)d_in[3]; const float* b1 = (const float*)d_in[4];
  const float* W2 = (const float*)d_in[5]; const float* b2 = (const float*)d_in[6];
  const float* W3 = (const float*)d_in[7]; const float* b3 = (const float*)d_in[8];
  const float* W4 = (const float*)d_in[9]; const float* b4 = (const float*)d_in[10];
  float* out = (float*)d_out;

  unsigned int* frags = (unsigned int*)d_ws;                 // 13*256 u32 = 13 KB

  prep_kernel<<<13, 256, 0, stream>>>(W1, b1, W2, b2, W3, b3, W4, b4, frags);
  init_kernel<<<(N3D_ + 255) / 256, 256, 0, stream>>>(out, N3D_);
  mlp_scatter_max_kernel<<<512, 256, 0, stream>>>(x, tidx, frags, out);
}
